// GraphSage_54065048323043
// MI455X (gfx1250) — compile-verified, matrix operand order checked by reading, not run on hardware
//
#include <hip/hip_runtime.h>

typedef __attribute__((ext_vector_type(16))) _Float16 v16h;
typedef __attribute__((ext_vector_type(8)))  _Float16 v8h;
typedef __attribute__((ext_vector_type(8)))  float    v8f;

#define N_NODES 50000
#define TILES   (N_NODES / 16)   // 3125, exact

// ---------------------------------------------------------------- utilities

__global__ void zero_f32(float* __restrict__ p, int n) {
    int i = blockIdx.x * blockDim.x + threadIdx.x;
    if (i < n) p[i] = 0.0f;
}

// one thread per edge: deg[dst] += 1
__global__ void degree_kernel(const int* __restrict__ dst,
                              float* __restrict__ deg, int E) {
    int e = blockIdx.x * blockDim.x + threadIdx.x;
    if (e < E) atomicAdd(&deg[dst[e]], 1.0f);
}

// one wave (32 lanes) per edge; each lane covers features {lane, lane+32}
__global__ void scatter_kernel(const float* __restrict__ feat,
                               const int* __restrict__ src,
                               const int* __restrict__ dst,
                               float* __restrict__ agg, int E) {
    int t = blockIdx.x * blockDim.x + threadIdx.x;
    int e = t >> 5;
    if (e >= E) return;
    int lane = t & 31;
    int s = src[e];
    int d = dst[e];
    const float* xp = feat + (size_t)s * 64;
    float*       ap = agg  + (size_t)d * 64;
    atomicAdd(ap + lane,      xp[lane]);
    atomicAdd(ap + lane + 32, xp[lane + 32]);
}

// ------------------------------------------------------- WMMA fragment I/O
// 16-bit A 16x32 layout (wave32):
//   lanes 0-15 : M = lane,    VGPR0..3 -> K=0..7,  VGPR4..7 -> K=16..23
//   lanes 16-31: M = lane-16, VGPR0..3 -> K=8..15, VGPR4..7 -> K=24..31
// kbase below already includes the per-half-wave K offset (koff).

__device__ inline v16h load_a_frag(const float* __restrict__ A, int row,
                                   int kbase, float scale) {
    v16h a;
    const float* p = A + (size_t)row * 64 + kbase;
#pragma unroll
    for (int i = 0; i < 8; ++i) a[i]     = (_Float16)(p[i]      * scale);
#pragma unroll
    for (int i = 0; i < 8; ++i) a[8 + i] = (_Float16)(p[16 + i] * scale);
    return a;
}

// B fragment from LDS-resident transposed f16 weights Wt[col][64]:
// lane holds column (lane&15); K indexing mirrors the A fragment.
// p points at Wt[col] + kbase; elements [0..7] and [16..23] -> two b128 loads.
__device__ inline v16h ld_bfrag(const _Float16* p) {
    v8h lo = *(const v8h*)(p);
    v8h hi = *(const v8h*)(p + 16);
    return __builtin_shufflevector(lo, hi, 0, 1, 2, 3, 4, 5, 6, 7,
                                           8, 9, 10, 11, 12, 13, 14, 15);
}

// ---------------------------------------------------------------- SAGE GEMM
// out[16xNOUT tile] = act( (agg*inv_deg)·Wn + xin·Wr + bias ), one wave/tile.
template <int NOUT, bool RELU>
__global__ void sage_gemm(const float* __restrict__ agg,
                          const float* __restrict__ xin,
                          const float* __restrict__ deg,
                          const float* __restrict__ Wn,
                          const float* __restrict__ Wr,
                          const float* __restrict__ bias,
                          float* __restrict__ out) {
    // Stage both weight matrices in LDS, transposed ([col][k]) and in f16.
    __shared__ __align__(16) _Float16 Wt[2][NOUT][64];
    for (int i = threadIdx.x; i < 64 * NOUT; i += blockDim.x) {
        int k = i / NOUT;            // source W is [64][NOUT] row-major
        int c = i - k * NOUT;
        Wt[0][c][k] = (_Float16)Wn[i];
        Wt[1][c][k] = (_Float16)Wr[i];
    }
    __syncthreads();

    int wave = (blockIdx.x * blockDim.x + threadIdx.x) >> 5;
    if (wave >= TILES) return;                 // wave-uniform: EXEC all-1 inside
    int lane = threadIdx.x & 31;
    int half = (lane >> 4) & 1;
    int koff = half * 8;
    int n    = lane & 15;
    int row  = wave * 16 + n;

    float dg    = deg[row];
    float scale = dg > 0.0f ? 1.0f / dg : 0.0f;

    // A fragments are shared across all output-column tiles -> hoist.
    v16h aN0 = load_a_frag(agg, row, 0 + koff,  scale);
    v16h aN1 = load_a_frag(agg, row, 32 + koff, scale);
    v16h aR0 = load_a_frag(xin, row, 0 + koff,  1.0f);
    v16h aR1 = load_a_frag(xin, row, 32 + koff, 1.0f);

#pragma unroll
    for (int nt = 0; nt < NOUT / 16; ++nt) {
        int col = nt * 16 + n;
        const _Float16* pn = &Wt[0][col][0];
        const _Float16* pr = &Wt[1][col][0];
        v8f c = {};
        c = __builtin_amdgcn_wmma_f32_16x16x32_f16(false, aN0, false, ld_bfrag(pn + koff),      (short)0, c, false, false);
        c = __builtin_amdgcn_wmma_f32_16x16x32_f16(false, aN1, false, ld_bfrag(pn + 32 + koff), (short)0, c, false, false);
        c = __builtin_amdgcn_wmma_f32_16x16x32_f16(false, aR0, false, ld_bfrag(pr + koff),      (short)0, c, false, false);
        c = __builtin_amdgcn_wmma_f32_16x16x32_f16(false, aR1, false, ld_bfrag(pr + 32 + koff), (short)0, c, false, false);

        float bv = bias[col];
        // C/D layout: VGPR r -> M = r + 8*half (lanes 0-15 vs 16-31), N = lane&15
#pragma unroll
        for (int r = 0; r < 8; ++r) {
            float v = c[r] + bv;
            if (RELU) v = v > 0.0f ? v : 0.0f;
            out[(size_t)(wave * 16 + r + half * 8) * NOUT + col] = v;
        }
    }
}

// ---------------------------------------------------------------- launcher

extern "C" void kernel_launch(void* const* d_in, const int* in_sizes, int n_in,
                              void* d_out, int out_size, void* d_ws, size_t ws_size,
                              hipStream_t stream) {
    const float* x   = (const float*)d_in[0];
    const int*   ei  = (const int*)  d_in[1];
    const float* W1n = (const float*)d_in[2];
    const float* W1r = (const float*)d_in[3];
    const float* b1  = (const float*)d_in[4];
    const float* W2n = (const float*)d_in[5];
    const float* W2r = (const float*)d_in[6];
    const float* b2  = (const float*)d_in[7];
    float* out = (float*)d_out;

    const int E = in_sizes[1] / 2;          // edge_index is [2, E]
    const int* src = ei;
    const int* dst = ei + E;

    // workspace: deg[N] | agg[N*64] | h[N*64]
    float* deg = (float*)d_ws;
    float* agg = deg + N_NODES;
    float* h   = agg + (size_t)N_NODES * 64;

    const int B = 256;

    // zero deg + agg (contiguous: N*65 floats)
    int zn = N_NODES * 65;
    zero_f32<<<(zn + B - 1) / B, B, 0, stream>>>(deg, zn);

    // degree once (same graph both layers)
    degree_kernel<<<(E + B - 1) / B, B, 0, stream>>>(dst, deg, E);

    // layer 1: scatter x -> agg, then fused mean+GEMM+bias+relu -> h
    long long st = (long long)E * 32;
    scatter_kernel<<<(int)((st + B - 1) / B), B, 0, stream>>>(x, src, dst, agg, E);

    int gw = (TILES * 32 + B - 1) / B;
    sage_gemm<64, true><<<gw, B, 0, stream>>>(agg, x, deg, W1n, W1r, b1, h);

    // layer 2: re-zero agg, scatter h, fused mean+GEMM+bias -> out
    zn = N_NODES * 64;
    zero_f32<<<(zn + B - 1) / B, B, 0, stream>>>(agg, zn);
    scatter_kernel<<<(int)((st + B - 1) / B), B, 0, stream>>>(h, src, dst, agg, E);
    sage_gemm<32, false><<<gw, B, 0, stream>>>(agg, h, deg, W2n, W2r, b2, out);
}
